// MoELayer_56478819942539
// MI455X (gfx1250) — matrix operand order repaired and checked by reading, not compiled, not run on hardware
//
#include <hip/hip_runtime.h>

// ---------------- problem constants (match reference) ----------------
#define DM   1024          // model dim D
#define NE   8             // experts
#define KT   2             // top-k
#define HH   2816          // ffn hidden
#define ATOK 8192          // B*S tokens
#define NA   (ATOK*KT)     // 16384 assignments
#define CAPE 2560          // capacity per expert

// ---------------- vector types for WMMA ----------------
typedef __attribute__((ext_vector_type(16))) __bf16 v16bf;
typedef __attribute__((ext_vector_type(8)))  __bf16 v8bf;
typedef __attribute__((ext_vector_type(8)))  float  v8f;
typedef __attribute__((ext_vector_type(4)))  int    v4i;

// ---------------- gfx1250 async global->LDS copy (guarded) ----------------
#if defined(__has_builtin)
#if __has_builtin(__builtin_amdgcn_global_load_async_to_lds_b128)
#define USE_ASYNC_LDS 1
#endif
#endif

typedef __attribute__((address_space(1))) v4i as1_v4i;
typedef __attribute__((address_space(3))) v4i as3_v4i;

static __device__ __forceinline__ void cp16_to_lds(void* l, const void* g) {
#ifdef USE_ASYNC_LDS
    // flat LDS pointer truncates to the LDS byte offset (ISA aperture rule);
    // integer casts also strip const / address-space qualifiers.
    __builtin_amdgcn_global_load_async_to_lds_b128(
        (as1_v4i*)(unsigned long long)(size_t)g,
        (as3_v4i*)(unsigned)(size_t)l, 0, 0);
#else
    *(uint4*)l = *(const uint4*)g;
#endif
}

#ifdef USE_ASYNC_LDS
#if __has_builtin(__builtin_amdgcn_s_wait_asynccnt)
#define WAIT_ASYNC() __builtin_amdgcn_s_wait_asynccnt(0)
#else
#define WAIT_ASYNC() asm volatile("s_wait_asynccnt 0x0" ::: "memory")
#endif
#else
#define WAIT_ASYNC() ((void)0)
#endif

static __device__ __forceinline__ unsigned short f2bf(float f) {
    unsigned int u = __float_as_uint(f);
    unsigned int r = (u + 0x7FFFu + ((u >> 16) & 1u)) >> 16;   // RNE
    return (unsigned short)r;
}

// A operand (16x32 bf16, MxK) fragment from two 16B chunks
static __device__ __forceinline__ v16bf mk_a(v8bf lo, v8bf hi) {
    return __builtin_shufflevector(lo, hi, 0,1,2,3,4,5,6,7,8,9,10,11,12,13,14,15);
}

// ---------------- 1) fp32 -> bf16 cast (weights) ----------------
__global__ void k_cast(const float* __restrict__ src, unsigned short* __restrict__ dst, int n4) {
    int i = blockIdx.x * 256 + threadIdx.x;
    if (i >= n4) return;
    float4 v = ((const float4*)src)[i];
    union { unsigned short u[4]; uint2 d; } o;
    o.u[0] = f2bf(v.x); o.u[1] = f2bf(v.y); o.u[2] = f2bf(v.z); o.u[3] = f2bf(v.w);
    ((uint2*)dst)[i] = o.d;
}

// ---------------- 2) router: logits, softmax, top-2, renorm ----------------
__global__ void __launch_bounds__(256) k_router(const float* __restrict__ x,
                                                const float* __restrict__ wr,
                                                int* __restrict__ top_e,
                                                float* __restrict__ top_w) {
    __shared__ float lw[NE * DM];                       // 32 KB of LDS
    for (int i = threadIdx.x; i < NE * DM; i += 256) lw[i] = wr[i];
    __syncthreads();

    const int lane = threadIdx.x & 31;
    const int wv   = threadIdx.x >> 5;
    const int t    = blockIdx.x * 8 + wv;               // one wave32 per token
    const float* xr = x + (size_t)t * DM;

    float acc[NE];
#pragma unroll
    for (int e = 0; e < NE; ++e) acc[e] = 0.f;

    for (int i = 0; i < DM / 32; ++i) {
        int d = i * 32 + lane;
        float xv = xr[d];
#pragma unroll
        for (int e = 0; e < NE; ++e) acc[e] += xv * lw[e * DM + d];
    }
#pragma unroll
    for (int off = 16; off >= 1; off >>= 1) {
#pragma unroll
        for (int e = 0; e < NE; ++e) acc[e] += __shfl_xor(acc[e], off, 32);
    }
    if (lane == 0) {
        float m = acc[0];
#pragma unroll
        for (int e = 1; e < NE; ++e) m = fmaxf(m, acc[e]);
        float p[NE];
#pragma unroll
        for (int e = 0; e < NE; ++e) p[e] = __expf(acc[e] - m);
        int e0 = 0;
#pragma unroll
        for (int e = 1; e < NE; ++e) if (p[e] > p[e0]) e0 = e;
        int e1 = (e0 == 0) ? 1 : 0;
#pragma unroll
        for (int e = 0; e < NE; ++e) if (e != e0 && p[e] > p[e1]) e1 = e;
        float s = p[e0] + p[e1];
        top_e[t * 2 + 0] = e0;  top_w[t * 2 + 0] = p[e0] / s;
        top_e[t * 2 + 1] = e1;  top_w[t * 2 + 1] = p[e1] / s;
    }
}

// ---------------- 3) stable per-expert rank (single workgroup, ballots) ----------------
__global__ void __launch_bounds__(1024) k_rank(const int* __restrict__ top_e,
                                               int* __restrict__ pos) {
    __shared__ int whist[32][NE];
    __shared__ int base[NE];
    if (threadIdx.x < NE) base[threadIdx.x] = 0;
    __syncthreads();

    const int lane = threadIdx.x & 31;
    const int wv   = threadIdx.x >> 5;
    const unsigned below = (lane == 0) ? 0u : (0xFFFFFFFFu >> (32 - lane));

    for (int r = 0; r < NA / 1024; ++r) {
        int a = r * 1024 + threadIdx.x;
        int e = top_e[a];
        unsigned mymask = 0;
#pragma unroll
        for (int q = 0; q < NE; ++q) {
            unsigned m = (unsigned)__ballot(e == q);
            if (q == e) mymask = m;
            if (lane == q) whist[wv][q] = __popc(m);
        }
        __syncthreads();
        int before = __popc(mymask & below);
        for (int w2 = 0; w2 < wv; ++w2) before += whist[w2][e];
        pos[a] = base[e] + before;
        __syncthreads();
        if (threadIdx.x < NE) {
            int s = 0;
            for (int w2 = 0; w2 < 32; ++w2) s += whist[w2][threadIdx.x];
            base[threadIdx.x] += s;
        }
        __syncthreads();
    }
}

// ---------------- 4) scatter token rows (fp32 -> bf16) into expert slots ----------------
__global__ void __launch_bounds__(256) k_scatter(const float* __restrict__ x,
                                                 const int* __restrict__ top_e,
                                                 const int* __restrict__ pos,
                                                 unsigned short* __restrict__ expert_xb) {
    const int a    = blockIdx.x * 8 + (threadIdx.x >> 5);   // one wave per assignment
    const int lane = threadIdx.x & 31;
    int p = pos[a];
    if (p >= CAPE) return;                                  // dropped by capacity
    int e = top_e[a];
    int t = a >> 1;                                         // K = 2
    const float4* src = (const float4*)(x + (size_t)t * DM);
    uint2* dst = (uint2*)(expert_xb + ((size_t)e * CAPE + p) * DM);
#pragma unroll
    for (int i = 0; i < 8; ++i) {
        int idx = i * 32 + lane;
        float4 v = src[idx];
        union { unsigned short u[4]; uint2 d; } o;
        o.u[0] = f2bf(v.x); o.u[1] = f2bf(v.y); o.u[2] = f2bf(v.z); o.u[3] = f2bf(v.w);
        dst[idx] = o.d;
    }
}

// ---------------- 5) FFN1: hidden = silu(x@w1^T) * (x@w3^T), bf16 out ----------------
// block 128 thr (4 waves); block tile: 32(M) x 128(N); wave tile: 32x32 for both mats.
// A (32x32-per-step) staged in LDS via async copy, double-buffered, padded rows (80B).
__global__ void __launch_bounds__(128) k_ffn1(const unsigned short* __restrict__ xb,
                                              const unsigned short* __restrict__ w1b,
                                              const unsigned short* __restrict__ w3b,
                                              unsigned short* __restrict__ hidb) {
    __shared__ __align__(16) unsigned char sA[2][32 * 80];

    const int e    = blockIdx.z;
    const int m0   = blockIdx.x * 32;
    const int lane = threadIdx.x & 31;
    const int wv   = threadIdx.x >> 5;
    const int n0   = blockIdx.y * 128 + wv * 32;
    const int half = lane >> 4;
    const int lm   = lane & 15;

    const __bf16* Abase = (const __bf16*)xb + ((size_t)e * CAPE + m0) * DM;
    const __bf16* B1    = (const __bf16*)w1b + (size_t)e * HH * DM;
    const __bf16* B3    = (const __bf16*)w3b + (size_t)e * HH * DM;

    // per-thread copy mapping: 128 threads x 16B = 32 rows x 64B
    const int crow = threadIdx.x >> 2;
    const int cch  = threadIdx.x & 3;
    const __bf16* gsrc = Abase + (size_t)crow * DM + cch * 8;

    v8f acc1[2][2] = {}, acc3[2][2] = {};

    cp16_to_lds(&sA[0][crow * 80 + cch * 16], gsrc);        // prologue: k-step 0

    for (int kk = 0; kk < DM / 32; ++kk) {
        const int k0  = kk * 32;
        const int buf = kk & 1;
        WAIT_ASYNC();
        __syncthreads();            // copies of step kk visible; step kk-1 readers done
        if (kk + 1 < DM / 32)
            cp16_to_lds(&sA[buf ^ 1][crow * 80 + cch * 16], gsrc + k0 + 32);

        v16bf av[2];
#pragma unroll
        for (int mt = 0; mt < 2; ++mt) {
            const unsigned char* rb = &sA[buf][(mt * 16 + lm) * 80];
            v8bf lo = *(const v8bf*)(rb + half * 32);
            v8bf hi = *(const v8bf*)(rb + half * 32 + 16);
            av[mt] = mk_a(lo, hi);
        }
#pragma unroll
        for (int nt = 0; nt < 2; ++nt) {
            size_t ncol = (size_t)(n0 + nt * 16 + lm);
            v16bf b1 = *(const v16bf*)(B1 + ncol * DM + k0 + half * 16);
            v16bf b3 = *(const v16bf*)(B3 + ncol * DM + k0 + half * 16);
#pragma unroll
            for (int mt = 0; mt < 2; ++mt) {
                acc1[mt][nt] = __builtin_amdgcn_wmma_f32_16x16x32_bf16(
                    false, av[mt], false, b1, (short)0, acc1[mt][nt], false, false);
                acc3[mt][nt] = __builtin_amdgcn_wmma_f32_16x16x32_bf16(
                    false, av[mt], false, b3, (short)0, acc3[mt][nt], false, false);
            }
        }
    }
    // fused SwiGLU epilogue; C layout: VGPR r, lane L -> (M = r + 8*half, N = L&15)
    unsigned short* out = hidb + (size_t)e * CAPE * HH;
#pragma unroll
    for (int mt = 0; mt < 2; ++mt)
#pragma unroll
        for (int nt = 0; nt < 2; ++nt)
#pragma unroll
            for (int r = 0; r < 8; ++r) {
                float h1 = acc1[mt][nt][r], h3 = acc3[mt][nt][r];
                float hv = (h1 / (1.f + __expf(-h1))) * h3;
                int m = m0 + mt * 16 + r + 8 * half;
                int n = n0 + nt * 16 + lm;
                out[(size_t)m * HH + n] = f2bf(hv);
            }
}

// ---------------- 6) FFN2: y = hidden @ w2^T, fp32 out ----------------
// block 128 thr (4 waves); block tile: 32(M) x 128(N); wave tile: 32x32.
__global__ void __launch_bounds__(128) k_ffn2(const unsigned short* __restrict__ hidb,
                                              const unsigned short* __restrict__ w2b,
                                              float* __restrict__ y) {
    __shared__ __align__(16) unsigned char sA[2][32 * 80];

    const int e    = blockIdx.z;
    const int m0   = blockIdx.x * 32;
    const int lane = threadIdx.x & 31;
    const int wv   = threadIdx.x >> 5;
    const int n0   = blockIdx.y * 128 + wv * 32;
    const int half = lane >> 4;
    const int lm   = lane & 15;

    const __bf16* Abase = (const __bf16*)hidb + ((size_t)e * CAPE + m0) * HH;
    const __bf16* B     = (const __bf16*)w2b  + (size_t)e * DM * HH;   // w2[e][d][h]

    const int crow = threadIdx.x >> 2;
    const int cch  = threadIdx.x & 3;
    const __bf16* gsrc = Abase + (size_t)crow * HH + cch * 8;

    v8f acc[2][2] = {};

    cp16_to_lds(&sA[0][crow * 80 + cch * 16], gsrc);

    for (int kk = 0; kk < HH / 32; ++kk) {
        const int k0  = kk * 32;
        const int buf = kk & 1;
        WAIT_ASYNC();
        __syncthreads();
        if (kk + 1 < HH / 32)
            cp16_to_lds(&sA[buf ^ 1][crow * 80 + cch * 16], gsrc + k0 + 32);

        v16bf av[2];
#pragma unroll
        for (int mt = 0; mt < 2; ++mt) {
            const unsigned char* rb = &sA[buf][(mt * 16 + lm) * 80];
            v8bf lo = *(const v8bf*)(rb + half * 32);
            v8bf hi = *(const v8bf*)(rb + half * 32 + 16);
            av[mt] = mk_a(lo, hi);
        }
#pragma unroll
        for (int nt = 0; nt < 2; ++nt) {
            size_t ncol = (size_t)(n0 + nt * 16 + lm);
            v16bf bv = *(const v16bf*)(B + ncol * HH + k0 + half * 16);
#pragma unroll
            for (int mt = 0; mt < 2; ++mt) {
                acc[mt][nt] = __builtin_amdgcn_wmma_f32_16x16x32_bf16(
                    false, av[mt], false, bv, (short)0, acc[mt][nt], false, false);
            }
        }
    }
    float* out = y + (size_t)e * CAPE * DM;
#pragma unroll
    for (int mt = 0; mt < 2; ++mt)
#pragma unroll
        for (int nt = 0; nt < 2; ++nt)
#pragma unroll
            for (int r = 0; r < 8; ++r) {
                int m = m0 + mt * 16 + r + 8 * half;
                int n = n0 + nt * 16 + lm;
                out[(size_t)m * DM + n] = acc[mt][nt][r];
            }
}

// ---------------- 7) combine: gather per-token, apply renormalized gate ----------------
__global__ void __launch_bounds__(256) k_combine(const float* __restrict__ y,
                                                 const int* __restrict__ top_e,
                                                 const float* __restrict__ top_w,
                                                 const int* __restrict__ pos,
                                                 float* __restrict__ out) {
    const int t  = blockIdx.x;
    const int d4 = threadIdx.x;
    float4 acc = make_float4(0.f, 0.f, 0.f, 0.f);
#pragma unroll
    for (int k = 0; k < KT; ++k) {
        int a = t * KT + k;
        int p = pos[a];
        if (p < CAPE) {
            int e = top_e[a];
            float w = top_w[a];
            float4 v = ((const float4*)(y + ((size_t)e * CAPE + p) * DM))[d4];
            acc.x += w * v.x; acc.y += w * v.y; acc.z += w * v.z; acc.w += w * v.w;
        }
    }
    ((float4*)(out + (size_t)t * DM))[d4] = acc;
}

// ---------------- host launcher ----------------
extern "C" void kernel_launch(void* const* d_in, const int* in_sizes, int n_in,
                              void* d_out, int out_size, void* d_ws, size_t ws_size,
                              hipStream_t stream) {
    const float* x  = (const float*)d_in[0];   // [B,S,D]
    const float* wr = (const float*)d_in[1];   // [E,D]
    const float* w1 = (const float*)d_in[2];   // [E,H,D]
    const float* w3 = (const float*)d_in[3];   // [E,H,D]
    const float* w2 = (const float*)d_in[4];   // [E,D,H]
    float* out = (float*)d_out;

    char* ws = (char*)d_ws;
    size_t off = 0;
    auto alloc = [&](size_t bytes) -> void* {
        void* p = ws + off;
        off = (off + bytes + 255) & ~(size_t)255;
        return p;
    };
    const size_t NW = (size_t)NE * HH * DM;
    unsigned short* w1b = (unsigned short*)alloc(NW * 2);
    unsigned short* w3b = (unsigned short*)alloc(NW * 2);
    unsigned short* w2b = (unsigned short*)alloc(NW * 2);
    unsigned short* exb = (unsigned short*)alloc((size_t)NE * CAPE * DM * 2);
    unsigned short* hid = (unsigned short*)alloc((size_t)NE * CAPE * HH * 2);
    float*          yb  = (float*)alloc((size_t)NE * CAPE * DM * 4);
    int*            t_e = (int*)alloc((size_t)NA * 4);
    float*          t_w = (float*)alloc((size_t)NA * 4);
    int*            pp  = (int*)alloc((size_t)NA * 4);

    {
        int n4 = (int)(NW / 4);
        int gs = (n4 + 255) / 256;
        k_cast<<<gs, 256, 0, stream>>>(w1, w1b, n4);
        k_cast<<<gs, 256, 0, stream>>>(w3, w3b, n4);
        k_cast<<<gs, 256, 0, stream>>>(w2, w2b, n4);
    }
    k_router<<<ATOK / 8, 256, 0, stream>>>(x, wr, t_e, t_w);
    k_rank<<<1, 1024, 0, stream>>>(t_e, pp);
    k_scatter<<<NA / 8, 256, 0, stream>>>(x, t_e, pp, exb);
    k_ffn1<<<dim3(CAPE / 32, HH / 128, NE), 128, 0, stream>>>(exb, w1b, w3b, hid);
    k_ffn2<<<dim3(CAPE / 32, DM / 128, NE), 128, 0, stream>>>(hid, w2b, yb);
    k_combine<<<ATOK, 256, 0, stream>>>(yb, t_e, t_w, pp, out);
}
